// HieraFormer_74680891343417
// MI455X (gfx1250) — matrix operand order repaired and checked by reading, not compiled, not run on hardware
//
#include <hip/hip_runtime.h>
#include <hip/hip_bf16.h>

// ---------------------------------------------------------------------------
// HieraFormer block for MI455X (gfx1250, wave32, WMMA).
//  B=2, C=512, N=2048, H=8, DK=64
//  out = x^T_res + LayerNorm( entmax15(q k^T / sqrt(DK)) v )
// bf16 WMMA (f32 accum) for all GEMMs; entmax via 30-step bisection on tau
// (equivalent to the sort-based exact algorithm to ~2^-31 in tau),
// renormalized by the computed sum. All entmax / LN math in f32.
// WMMA loops are software-pipelined with unroll-by-2 ping-pong buffers:
// loads of step i+1 overlap WMMAs of step i, with no register copies.
// ---------------------------------------------------------------------------

typedef __bf16 bf16;
typedef __attribute__((ext_vector_type(8)))  float v8f;
typedef __attribute__((ext_vector_type(16))) bf16  v16bf;
typedef __attribute__((ext_vector_type(8)))  bf16  v8bf;

#define DEV static __device__ __forceinline__

constexpr int B_  = 2;
constexpr int C_  = 512;
constexpr int N_  = 2048;
constexpr int H_  = 8;
constexpr int DK_ = 64;
constexpr int BN  = B_ * N_;          // 4096 tokens
constexpr int SSTR = 2052;            // padded LDS stride for energy rows
constexpr int OSTR = 68;              // padded LDS stride for O partials

DEV v8f zero8() {
  v8f z;
  #pragma unroll
  for (int i = 0; i < 8; ++i) z[i] = 0.0f;
  return z;
}

DEV float wave_sum(float v) {
  #pragma unroll
  for (int m = 16; m > 0; m >>= 1) v += __shfl_xor(v, m, 32);
  return v;
}
DEV float wave_max(float v) {
  #pragma unroll
  for (int m = 16; m > 0; m >>= 1) v = fmaxf(v, __shfl_xor(v, m, 32));
  return v;
}

DEV v16bf ld_a16(const bf16* p) {           // A-frag halves: [p..p+7], [p+16..p+23]
  v16bf a;
  v8bf lo = *(const v8bf*)p;
  v8bf hi = *(const v8bf*)(p + 16);
  #pragma unroll
  for (int i = 0; i < 8; ++i) { a[i] = lo[i]; a[i + 8] = hi[i]; }
  return a;
}
DEV v16bf ld_b16(const bf16* p) {           // B-frag halves: [p..p+7], [p+16..p+23]
  v16bf b;
  v8bf lo = *(const v8bf*)p;
  v8bf hi = *(const v8bf*)(p + 16);
  #pragma unroll
  for (int i = 0; i < 8; ++i) { b[i] = lo[i]; b[i + 8] = hi[i]; }
  return b;
}

// ---------------------------------------------------------------------------
// Kernel 0a: x (B,C,N) f32  ->  xb (B*N, C) bf16   (transpose + convert)
// ---------------------------------------------------------------------------
__global__ void k_prep_x(const float* __restrict__ x, bf16* __restrict__ xb) {
  int idx = blockIdx.x * 256 + threadIdx.x;       // over BN*C
  int c   = idx & (C_ - 1);
  int tok = idx >> 9;                             // b*N + n
  int b   = tok >> 11;
  int n   = tok & (N_ - 1);
  xb[idx] = (bf16)x[((size_t)b * C_ + c) * N_ + n];
}

// ---------------------------------------------------------------------------
// Kernel 0b: Wq|Wk|Wv (C,C) f32 -> wb (3,C,C) bf16
// ---------------------------------------------------------------------------
__global__ void k_prep_w(const float* __restrict__ wq, const float* __restrict__ wk,
                         const float* __restrict__ wv, bf16* __restrict__ wb) {
  int idx = blockIdx.x * 256 + threadIdx.x;       // over 3*C*C
  int m   = idx / (C_ * C_);
  int r   = idx - m * (C_ * C_);
  const float* src = (m == 0) ? wq : (m == 1) ? wk : wv;
  wb[idx] = (bf16)src[r];
}

// ---------------------------------------------------------------------------
// Kernel 1: QKV projection GEMM.  y = xb @ W^T + bias  (M=4096, K=512, N=512)
// grid (BN/16, 3), block 256 (8 waves). Wave w owns output cols [64w,64w+64).
// Unroll-by-2 ping-pong: loads for step i+1 overlap WMMAs of step i.
// q,k stored (B,H,N,DK) bf16; v stored transposed (B,H,DK,N) bf16.
// ---------------------------------------------------------------------------
__global__ __launch_bounds__(256, 1)
void k_qkv(const bf16* __restrict__ xb, const bf16* __restrict__ wb,
           const float* __restrict__ bq, const float* __restrict__ bk,
           const float* __restrict__ bv,
           bf16* __restrict__ qb, bf16* __restrict__ kb, bf16* __restrict__ vtb) {
  const int z     = blockIdx.y;                   // 0=q 1=k 2=v
  const int m0    = blockIdx.x * 16;              // token tile
  const int lane  = threadIdx.x & 31;
  const int wid   = threadIdx.x >> 5;
  const int mrow  = lane & 15;
  const int khalf = lane >> 4;

  const bf16*  W    = wb + (size_t)z * C_ * C_;
  const float* bias = (z == 0) ? bq : (z == 1) ? bk : bv;

  const bf16* arow = xb + (size_t)(m0 + mrow) * C_ + khalf * 8;
  const bf16* wrow[4];
  #pragma unroll
  for (int t = 0; t < 4; ++t)
    wrow[t] = W + (size_t)(wid * 64 + t * 16 + mrow) * C_ + khalf * 8;

  v8f acc[4];
  #pragma unroll
  for (int t = 0; t < 4; ++t) acc[t] = zero8();

  // prologue: buffer0 holds fragments for kk = 0
  v16bf a0 = ld_a16(arow);
  v16bf b0[4];
  #pragma unroll
  for (int t = 0; t < 4; ++t) b0[t] = ld_b16(wrow[t]);

  for (int kk = 0; kk < C_; kk += 64) {
    const int k1 = kk + 32;                       // always < C_
    const int k2 = (kk + 64) & (C_ - 1);          // wraps to 0 on last iter

    v16bf a1 = ld_a16(arow + k1);                 // fetch k1 ...
    v16bf b1[4];
    #pragma unroll
    for (int t = 0; t < 4; ++t) b1[t] = ld_b16(wrow[t] + k1);
    #pragma unroll
    for (int t = 0; t < 4; ++t)                   // ... while multiplying kk
      acc[t] = __builtin_amdgcn_wmma_f32_16x16x32_bf16(
          false, a0, false, b0[t], (short)0, acc[t], false, false);

    a0 = ld_a16(arow + k2);                       // fetch k2 ...
    #pragma unroll
    for (int t = 0; t < 4; ++t) b0[t] = ld_b16(wrow[t] + k2);
    #pragma unroll
    for (int t = 0; t < 4; ++t)                   // ... while multiplying k1
      acc[t] = __builtin_amdgcn_wmma_f32_16x16x32_bf16(
          false, a1, false, b1[t], (short)0, acc[t], false, false);
  }

  #pragma unroll
  for (int t = 0; t < 4; ++t) {
    int   j  = wid * 64 + t * 16 + mrow;
    float bj = bias[j];
    int   h  = j >> 6;
    int   d  = j & 63;
    if (z < 2) {
      bf16* dst = (z == 0) ? qb : kb;
      #pragma unroll
      for (int r = 0; r < 8; ++r) {
        int tok = m0 + r + khalf * 8;
        int b   = tok >> 11, n = tok & (N_ - 1);
        dst[(((size_t)b * H_ + h) * N_ + n) * DK_ + d] = (bf16)(acc[t][r] + bj);
      }
    } else {                                      // v transposed: 8 contiguous tokens
      int tok0 = m0 + khalf * 8;
      int b    = tok0 >> 11, n0 = tok0 & (N_ - 1);
      v8bf pk;
      #pragma unroll
      for (int r = 0; r < 8; ++r) pk[r] = (bf16)(acc[t][r] + bj);
      *(v8bf*)(vtb + (((size_t)b * H_ + h) * DK_ + d) * N_ + n0) = pk;
    }
  }
}

// ---------------------------------------------------------------------------
// Kernel 2: attention. WG = 16 query rows of one (b,h). 256 threads (8 waves).
// LDS: S[16][SSTR] f32 energies (128KB) + OP[2][16][OSTR] f32 partials.
// (140KB/WG -> two workgroups co-resident per 320KB WGP.)
// ---------------------------------------------------------------------------
__global__ __launch_bounds__(256, 1)
void k_attn(const bf16* __restrict__ qb, const bf16* __restrict__ kb,
            const bf16* __restrict__ vtb, float* __restrict__ ob) {
  extern __shared__ float smem[];
  float* S  = smem;                 // 16 * SSTR
  float* OP = smem + 16 * SSTR;     // 2 * 16 * OSTR

  const int n0    = blockIdx.x * 16;
  const int h     = blockIdx.y;
  const int b     = blockIdx.z;
  const int lane  = threadIdx.x & 31;
  const int wid   = threadIdx.x >> 5;
  const int mrow  = lane & 15;
  const int khalf = lane >> 4;

  const size_t headQ = ((size_t)b * H_ + h) * N_ * DK_;
  const bf16* q  = qb  + headQ;
  const bf16* k  = kb  + headQ;
  const bf16* vt = vtb + ((size_t)b * H_ + h) * DK_ * N_;

  // ---- phase 1: energy S = (q k^T) * 0.0625  (1/sqrt(64) * 1/2 for entmax)
  {
    v16bf aq[2];
    #pragma unroll
    for (int j = 0; j < 2; ++j)
      aq[j] = ld_a16(q + (size_t)(n0 + mrow) * DK_ + j * 32 + khalf * 8);

    const bf16* kbase = k + (size_t)mrow * DK_ + khalf * 8;
    // wave wid handles key tiles kt = wid + 8*j, j = 0..15 (ping-pong by 2)
    auto koff = [&](int j) { return (size_t)((wid + 8 * (j & 15)) * 16) * DK_; };

    v16bf b0a = ld_b16(kbase + koff(0));
    v16bf b0b = ld_b16(kbase + koff(0) + 32);

    for (int j = 0; j < 16; j += 2) {
      const int ktA = wid + 8 * j;
      const int ktB = wid + 8 * (j + 1);

      v16bf b1a = ld_b16(kbase + koff(j + 1));    // fetch tile j+1 ...
      v16bf b1b = ld_b16(kbase + koff(j + 1) + 32);
      __builtin_prefetch(kbase + koff(j + 3), 0, 0);

      v8f c = zero8();                            // ... while multiplying tile j
      c = __builtin_amdgcn_wmma_f32_16x16x32_bf16(
          false, aq[0], false, b0a, (short)0, c, false, false);
      c = __builtin_amdgcn_wmma_f32_16x16x32_bf16(
          false, aq[1], false, b0b, (short)0, c, false, false);
      #pragma unroll
      for (int r = 0; r < 8; ++r)
        S[(r + khalf * 8) * SSTR + ktA * 16 + mrow] = c[r] * 0.0625f;

      b0a = ld_b16(kbase + koff(j + 2));          // fetch tile j+2 ...
      b0b = ld_b16(kbase + koff(j + 2) + 32);

      c = zero8();                                // ... while multiplying tile j+1
      c = __builtin_amdgcn_wmma_f32_16x16x32_bf16(
          false, aq[0], false, b1a, (short)0, c, false, false);
      c = __builtin_amdgcn_wmma_f32_16x16x32_bf16(
          false, aq[1], false, b1b, (short)0, c, false, false);
      #pragma unroll
      for (int r = 0; r < 8; ++r)
        S[(r + khalf * 8) * SSTR + ktB * 16 + mrow] = c[r] * 0.0625f;
    }
  }
  // warm the cache with this wave's V tiles while entmax runs
  __builtin_prefetch(vt + (size_t)((wid & 3) * 16 + mrow) * N_ + (wid >> 2) * 1024, 0, 0);
  __syncthreads();

  // ---- phase 2: exact 1.5-entmax per row via bisection on tau.
  // Each wave owns rows {2*wid, 2*wid+1}; lane caches its 64 scores in VGPRs.
  for (int rr = 0; rr < 2; ++rr) {
    const int row = wid * 2 + rr;
    float* Sr = S + row * SSTR;
    float zv[64];
    float mx = -3.4e38f;
    #pragma unroll
    for (int i = 0; i < 64; ++i) { zv[i] = Sr[lane + 32 * i]; mx = fmaxf(mx, zv[i]); }
    mx = wave_max(mx);
    #pragma unroll
    for (int i = 0; i < 64; ++i) zv[i] -= mx;     // max(z) == 0 now; tau* in [-1,0]

    float tlo = -1.0f, thi = 0.0f;
    for (int it = 0; it < 30; ++it) {
      float t = 0.5f * (tlo + thi);
      float s = 0.0f;
      #pragma unroll
      for (int i = 0; i < 64; ++i) {
        float d = fmaxf(zv[i] - t, 0.0f);
        s = fmaf(d, d, s);
      }
      s = wave_sum(s);
      if (s >= 1.0f) tlo = t; else thi = t;       // f(tau) decreasing, f(tau*)=1
    }
    const float tau = 0.5f * (tlo + thi);
    float s = 0.0f;
    #pragma unroll
    for (int i = 0; i < 64; ++i) {
      float d = fmaxf(zv[i] - tau, 0.0f);
      s = fmaf(d, d, s);
    }
    s = wave_sum(s);
    const float inv = 1.0f / s;                   // s ~= 1; renormalize exactly
    #pragma unroll
    for (int i = 0; i < 64; ++i) {
      float d = fmaxf(zv[i] - tau, 0.0f);
      Sr[lane + 32 * i] = d * d * inv;
    }
  }
  __syncthreads();

  // ---- phase 3: O = P @ V.  Wave wid -> d-tile (wid&3), key half (wid>>2).
  {
    const int dt   = (wid & 3) * 16;
    const int half = wid >> 2;
    const bf16*  vbase = vt + (size_t)(dt + mrow) * N_ + half * 1024 + khalf * 8;
    const float* pbase = S + mrow * SSTR + half * 1024 + khalf * 8;

    v8f acc = zero8();
    v16bf bv0 = ld_b16(vbase);                    // prologue: kt2 = 0
    for (int kt2 = 0; kt2 < 32; kt2 += 2) {
      v16bf bv1 = ld_b16(vbase + ((kt2 + 1) & 31) * 32);   // fetch kt2+1 ...
      __builtin_prefetch(vbase + ((kt2 + 3) & 31) * 32, 0, 0);

      v16bf ap;                                   // P fragment from LDS (f32->bf16)
      {
        const float* p = pbase + kt2 * 32;
        #pragma unroll
        for (int i = 0; i < 8; ++i) { ap[i] = (bf16)p[i]; ap[i + 8] = (bf16)p[i + 16]; }
      }
      acc = __builtin_amdgcn_wmma_f32_16x16x32_bf16(   // ... while multiplying kt2
          false, ap, false, bv0, (short)0, acc, false, false);

      bv0 = ld_b16(vbase + ((kt2 + 2) & 31) * 32);         // fetch kt2+2 ...
      {
        const float* p = pbase + (kt2 + 1) * 32;
        #pragma unroll
        for (int i = 0; i < 8; ++i) { ap[i] = (bf16)p[i]; ap[i + 8] = (bf16)p[i + 16]; }
      }
      acc = __builtin_amdgcn_wmma_f32_16x16x32_bf16(   // ... while multiplying kt2+1
          false, ap, false, bv1, (short)0, acc, false, false);
    }
    #pragma unroll
    for (int r = 0; r < 8; ++r)
      OP[half * (16 * OSTR) + (r + khalf * 8) * OSTR + dt + mrow] = acc[r];
  }
  __syncthreads();

  // ---- phase 4: reduce key-halves, write o (B,N,C) f32
  for (int i = threadIdx.x; i < 16 * 64; i += 256) {
    int m = i >> 6, d = i & 63;
    float val = OP[m * OSTR + d] + OP[16 * OSTR + m * OSTR + d];
    ob[((size_t)(b * N_ + n0 + m)) * C_ + h * DK_ + d] = val;
  }
}

// ---------------------------------------------------------------------------
// Kernel 3: out = x^T + LayerNorm(o)   (unbiased std, eps added to std)
// one block (256 thr) per token row of C=512.
// ---------------------------------------------------------------------------
__global__ __launch_bounds__(256, 1)
void k_ln(const float* __restrict__ ob, const float* __restrict__ x,
          const float* __restrict__ a2, const float* __restrict__ b2,
          float* __restrict__ out) {
  __shared__ float red[16];
  const int tok = blockIdx.x;
  const int b = tok >> 11, n = tok & (N_ - 1);
  const float* orow = ob + (size_t)tok * C_;

  float v0 = orow[threadIdx.x];
  float v1 = orow[threadIdx.x + 256];
  float s  = v0 + v1;
  float ss = v0 * v0 + v1 * v1;
  s  = wave_sum(s);
  ss = wave_sum(ss);
  const int wid = threadIdx.x >> 5, lane = threadIdx.x & 31;
  if (lane == 0) { red[wid] = s; red[8 + wid] = ss; }
  __syncthreads();
  s = 0.0f; ss = 0.0f;
  #pragma unroll
  for (int i = 0; i < 8; ++i) { s += red[i]; ss += red[8 + i]; }

  const float mean = s * (1.0f / C_);
  const float var  = fmaxf((ss - C_ * mean * mean) * (1.0f / (C_ - 1)), 0.0f);
  const float istd = 1.0f / (sqrtf(var) + 1e-6f);

  #pragma unroll
  for (int e = 0; e < 2; ++e) {
    int c = threadIdx.x + e * 256;
    float xr = x[((size_t)b * C_ + c) * N_ + n];
    out[(size_t)tok * C_ + c] = a2[c] * (orow[c] - mean) * istd + b2[c] + xr;
  }
}

// ---------------------------------------------------------------------------
extern "C" void kernel_launch(void* const* d_in, const int* in_sizes, int n_in,
                              void* d_out, int out_size, void* d_ws, size_t ws_size,
                              hipStream_t stream) {
  (void)in_sizes; (void)n_in; (void)out_size; (void)ws_size;
  const float* x   = (const float*)d_in[0];
  const float* Wq  = (const float*)d_in[1];
  const float* bq  = (const float*)d_in[2];
  const float* Wk  = (const float*)d_in[3];
  const float* bk  = (const float*)d_in[4];
  const float* Wv  = (const float*)d_in[5];
  const float* bv  = (const float*)d_in[6];
  const float* a2  = (const float*)d_in[7];
  const float* b2  = (const float*)d_in[8];
  float* out = (float*)d_out;

  // workspace carve-out (~25.5 MB total)
  char*  ws  = (char*)d_ws;
  bf16*  xb  = (bf16*)ws;                                   // BN*C bf16   (4 MB)
  size_t off = (size_t)BN * C_ * sizeof(bf16);
  bf16*  wb  = (bf16*)(ws + off);  off += 3ull * C_ * C_ * sizeof(bf16);   // 1.5 MB
  bf16*  qb  = (bf16*)(ws + off);  off += (size_t)BN * C_ * sizeof(bf16);  // 4 MB
  bf16*  kb  = (bf16*)(ws + off);  off += (size_t)BN * C_ * sizeof(bf16);  // 4 MB
  bf16*  vtb = (bf16*)(ws + off);  off += (size_t)BN * C_ * sizeof(bf16);  // 4 MB
  float* ob  = (float*)(ws + off);                                         // 8 MB

  k_prep_x<<<BN * C_ / 256, 256, 0, stream>>>(x, xb);
  k_prep_w<<<3 * C_ * C_ / 256, 256, 0, stream>>>(Wq, Wk, Wv, wb);
  k_qkv<<<dim3(BN / 16, 3), 256, 0, stream>>>(xb, wb, bq, bk, bv, qb, kb, vtb);

  size_t shmem = (16 * SSTR + 2 * 16 * OSTR) * sizeof(float);   // ~140 KB (< 320 KB/WGP)
  k_attn<<<dim3(N_ / 16, H_, B_), 256, shmem, stream>>>(qb, kb, vtb, ob);

  k_ln<<<BN, 256, 0, stream>>>(ob, x, a2, b2, out);
}
